// DynamicLocalPoolPointnet_13778255086350
// MI455X (gfx1250) — matrix operand order repaired and checked by reading, not compiled, not run on hardware
//
#include <hip/hip_runtime.h>
#include <hip/hip_bf16.h>

typedef __attribute__((ext_vector_type(16))) _Float16 v16h;
typedef __attribute__((ext_vector_type(8)))  _Float16 v8h;
typedef __attribute__((ext_vector_type(8)))  float    v8f;

#define B_    8
#define T_    32768
#define HID_  128
#define RESO_ 64
#define R2_   4096

// ---- workspace byte offsets -------------------------------------------------
#define OFF_HMAX  (1u<<20)                     // after 942KB of f16 weights
#define OFF_NETPL (OFF_HMAX + 4096u)
#define OFF_CMAT  (OFF_NETPL + 4096u)
#define OFF_IDX   (OFF_CMAT + 4096u)
#define OFF_NET   (OFF_IDX + 3145728u)
#define OFF_BINS  (OFF_NET + 134217728u)
#define OFF_CNT   (OFF_BINS + 50331648u)

// ---- swizzled f16 weight offsets (in halves) --------------------------------
#define W_PLN_IN 0
#define W_PLN_H1 4096
#define W_PLN_H2 20480
#define W_FC_POS 36864
#define W_FC0    45056      /* + i*32768 */
#define W_FC1    208896     /* + i*16384 */
#define W_SC     290816     /* + i*32768 */
#define W_FCC    454656

// ---- helpers ----------------------------------------------------------------
__device__ __forceinline__ v8f wmma32(v16h a, v16h b, v8f c) {
  return __builtin_amdgcn_wmma_f32_16x16x32_f16(false, a, false, b, (short)0, c, false, false);
}

// A fragment (16x32 f16) from LDS row-major activations, row stride = ldk halves.
// lanes 0-15: M=lane, K in {k0..k0+7, k0+16..k0+23}; lanes 16-31: M=lane-16, K+8.
__device__ __forceinline__ v16h load_afrag(const _Float16* lact, int ldk, int lane, int k0) {
  int m = lane & 15;
  int off = (lane >> 4) * 8;
  const _Float16* row = lact + m * ldk + k0 + off;
  v8h lo = *(const v8h*)(row);
  v8h hi = *(const v8h*)(row + 16);
  return __builtin_shufflevector(lo, hi, 0,1,2,3,4,5,6,7,8,9,10,11,12,13,14,15);
}

// B fragment (32x16 f16), pre-swizzled in global memory: 16 contiguous halves/lane.
__device__ __forceinline__ v16h load_bfrag(const _Float16* wbase, int ksteps, int ntile, int kstep, int lane) {
  return *(const v16h*)(wbase + (size_t)(((ntile * ksteps) + kstep) * 32 + lane) * 16);
}

__device__ __forceinline__ v16h frag_relu(v16h a) {
  v16h r;
  #pragma unroll
  for (int e = 0; e < 16; ++e) { _Float16 x = a[e]; r[e] = (x > (_Float16)0.0f) ? x : (_Float16)0.0f; }
  return r;
}

// monotonic float->uint key so atomicMax(u32) == float max (handles negatives)
__device__ __forceinline__ unsigned fkey(float f) {
  unsigned b = __float_as_uint(f);
  return (b & 0x80000000u) ? ~b : (b | 0x80000000u);
}
__device__ __forceinline__ float funkey(unsigned k) {
  unsigned b = (k & 0x80000000u) ? (k ^ 0x80000000u) : ~k;
  return __uint_as_float(b);
}

// ---- weight pre-swizzle: W (K x N f32 row-major) -> WMMA B fragments f16 ----
__global__ void k_swizzle(const float* __restrict__ W, _Float16* __restrict__ dst,
                          int K, int Kpad, int N) {
  int t = blockIdx.x * blockDim.x + threadIdx.x;
  int total = N * Kpad;
  if (t >= total) return;
  int e = t & 15;
  int lane = (t >> 4) & 31;
  int rest = t >> 9;                 // ntile*ksteps + kstep
  int ksteps = Kpad >> 5;
  int kstep = rest % ksteps;
  int ntile = rest / ksteps;
  int k = kstep * 32 + ((lane >= 16) ? 8 : 0) + (e & 7) + ((e >> 3) << 4);
  int n = ntile * 16 + (lane & 15);
  float v = (k < K) ? W[(size_t)k * N + n] : 0.0f;
  dst[t] = (_Float16)v;
}

__global__ void k_fill_f32(float* p, float v, long n) {
  long i = (long)blockIdx.x * blockDim.x + threadIdx.x;
  if (i < n) p[i] = v;
}
__global__ void k_fill_key(unsigned* p, float v, long n) {
  long i = (long)blockIdx.x * blockDim.x + threadIdx.x;
  if (i < n) p[i] = fkey(v);
}

// ---- pointnet MLP (3->128->128->128) + max-pool via atomic max --------------
__global__ void __launch_bounds__(256)
k_pointnet(const float* __restrict__ p,
           const float* __restrict__ b_in, const float* __restrict__ b_h1, const float* __restrict__ b_h2,
           const _Float16* __restrict__ w_in, const _Float16* __restrict__ w_h1, const _Float16* __restrict__ w_h2,
           float* __restrict__ hmax) {
  __shared__ __align__(32) _Float16 lact[8][16][128];
  const int wv = threadIdx.x >> 5, lane = threadIdx.x & 31;
  const int nl = lane & 15, mbase = (lane >> 4) * 8;
  const int pt0 = blockIdx.x * 128 + wv * 16;
  const int b = pt0 / T_;

  { // stage p padded to K=32 into columns [0,32) of each row
    int m = lane & 15, cb = (lane >> 4) * 16;
    const float* pp = p + (size_t)(pt0 + m) * 3;
    for (int c = 0; c < 16; ++c) {
      int k = cb + c;
      lact[wv][m][k] = (k < 3) ? (_Float16)pp[k] : (_Float16)0.0f;
    }
  }
  __syncthreads();
  v16h a0 = load_afrag(&lact[wv][0][0], 128, lane, 0);
  __syncthreads();
  // layer 1: K=32(pad), N=128
  for (int nt = 0; nt < 8; ++nt) {
    v8f acc = {};
    acc = wmma32(a0, load_bfrag(w_in, 1, nt, 0, lane), acc);
    float bias = b_in[nt * 16 + nl];
    #pragma unroll
    for (int r = 0; r < 8; ++r) {
      float v = acc[r] + bias; v = v > 0.f ? v : 0.f;
      lact[wv][mbase + r][nt * 16 + nl] = (_Float16)v;
    }
  }
  __syncthreads();
  v16h af[4];
  #pragma unroll
  for (int k = 0; k < 4; ++k) af[k] = load_afrag(&lact[wv][0][0], 128, lane, 32 * k);
  __syncthreads();
  // layer 2: K=128, N=128
  for (int nt = 0; nt < 8; ++nt) {
    v8f acc = {};
    #pragma unroll
    for (int k = 0; k < 4; ++k) acc = wmma32(af[k], load_bfrag(w_h1, 4, nt, k, lane), acc);
    float bias = b_h1[nt * 16 + nl];
    #pragma unroll
    for (int r = 0; r < 8; ++r) {
      float v = acc[r] + bias; v = v > 0.f ? v : 0.f;
      lact[wv][mbase + r][nt * 16 + nl] = (_Float16)v;
    }
  }
  __syncthreads();
  #pragma unroll
  for (int k = 0; k < 4; ++k) af[k] = load_afrag(&lact[wv][0][0], 128, lane, 32 * k);
  // layer 3 + per-column max + atomic max into hmax (relu>=0 so int compare ok)
  for (int nt = 0; nt < 8; ++nt) {
    v8f acc = {};
    #pragma unroll
    for (int k = 0; k < 4; ++k) acc = wmma32(af[k], load_bfrag(w_h2, 4, nt, k, lane), acc);
    float bias = b_h2[nt * 16 + nl];
    float mx = 0.f;
    #pragma unroll
    for (int r = 0; r < 8; ++r) {
      float v = acc[r] + bias; v = v > 0.f ? v : 0.f;
      mx = v > mx ? v : mx;
    }
    atomicMax((int*)&hmax[b * HID_ + nt * 16 + nl], __float_as_int(mx));
  }
}

// ---- per-batch head: h3, pln_out, change_basis, net_pl ----------------------
__global__ void __launch_bounds__(128)
k_head(const float* __restrict__ hmax,
       const float* __restrict__ w3, const float* __restrict__ b3,
       const float* __restrict__ wout, const float* __restrict__ bout,
       const float* __restrict__ wph, const float* __restrict__ bph,
       float* __restrict__ netpl, float* __restrict__ cmatws, float* __restrict__ outc) {
  __shared__ float hm[128], h3[128], raw[9];
  int b = blockIdx.x, n = threadIdx.x;
  hm[n] = hmax[b * 128 + n];
  __syncthreads();
  float s = b3[n];
  for (int k = 0; k < 128; ++k) s += hm[k] * w3[k * 128 + n];
  h3[n] = s > 0.f ? s : 0.f;
  __syncthreads();
  if (n < 9) {
    float t = bout[n];
    for (int k = 0; k < 128; ++k) t += h3[k] * wout[k * 9 + n];
    raw[n] = t;
  }
  __syncthreads();
  float o = bph[n];
  for (int j = 0; j < 9; ++j) {
    float rj = raw[j]; rj = rj > 0.f ? rj : 0.f;
    o += rj * wph[j * 128 + n];
  }
  netpl[b * 128 + n] = o;
  if (n < 3) {
    float ppx = raw[3*n], ppy = raw[3*n+1], ppz = raw[3*n+2];
    float nn = sqrtf(ppx*ppx + ppy*ppy + ppz*ppz);
    float nx = ppx/nn + 1e-6f, ny = ppy/nn + 1e-6f, nz = ppz/nn + 1e-6f;
    float S[3][3] = {{0.f, 0.f, nx}, {0.f, 0.f, ny}, {-nx, -ny, 0.f}};
    float S2[3][3];
    for (int i = 0; i < 3; ++i)
      for (int j = 0; j < 3; ++j) {
        float a = 0.f;
        for (int k = 0; k < 3; ++k) a += S[i][k] * S[k][j];
        S2[i][j] = a;
      }
    float dotv = 1.0f - nz, divv = nx*nx + ny*ny;
    float R[3][3];
    for (int i = 0; i < 3; ++i)
      for (int j = 0; j < 3; ++j)
        R[i][j] = ((i == j) ? 1.f : 0.f) + S[i][j] + S2[i][j] * dotv / divv;
    float det = R[0][0]*(R[1][1]*R[2][2]-R[1][2]*R[2][1])
              - R[0][1]*(R[1][0]*R[2][2]-R[1][2]*R[2][0])
              + R[0][2]*(R[1][0]*R[2][1]-R[1][1]*R[2][0]);
    float id = 1.0f / det;
    float CI[9];
    CI[0]=(R[1][1]*R[2][2]-R[1][2]*R[2][1])*id;
    CI[1]=(R[0][2]*R[2][1]-R[0][1]*R[2][2])*id;
    CI[2]=(R[0][1]*R[1][2]-R[0][2]*R[1][1])*id;
    CI[3]=(R[1][2]*R[2][0]-R[1][0]*R[2][2])*id;
    CI[4]=(R[0][0]*R[2][2]-R[0][2]*R[2][0])*id;
    CI[5]=(R[0][2]*R[1][0]-R[0][0]*R[1][2])*id;
    CI[6]=(R[1][0]*R[2][1]-R[1][1]*R[2][0])*id;
    CI[7]=(R[0][1]*R[2][0]-R[0][0]*R[2][1])*id;
    CI[8]=(R[0][0]*R[1][1]-R[0][1]*R[1][0])*id;
    float bx0=fabsf(R[0][0]), bx1=fabsf(R[1][0]), bx2=fabsf(R[2][0]);
    float by0=fabsf(R[0][1]), by1=fabsf(R[1][1]), by2=fabsf(R[2][1]);
    float sx2 = bx0*bx0+bx1*bx1+bx2*bx2, sy2 = by0*by0+by1*by1+by2*by2;
    float npx = ((bx0+bx1+bx2)/sx2) * sqrtf(sx2);
    float npy = ((by0+by1+by2)/sy2) * sqrtf(sy2);
    float nm = npx > npy ? npx : npy;
    float* d1 = cmatws + (b*3 + n) * 12;
    float* d2 = outc   + (b*3 + n) * 12;
    float vals[12] = {CI[0],CI[1],CI[2],CI[3],CI[4],CI[5],CI[6],CI[7],CI[8],nm,nm,nm};
    for (int i = 0; i < 12; ++i) { d1[i] = vals[i]; d2[i] = vals[i]; }
  }
}

// ---- per-point dynamic bin indices ------------------------------------------
__global__ void __launch_bounds__(256)
k_idx(const float* __restrict__ p, const float* __restrict__ cmat, int* __restrict__ idx) {
  int pt = blockIdx.x * blockDim.x + threadIdx.x;
  if (pt >= B_ * T_) return;
  int b = pt / T_;
  float x = p[(size_t)pt*3], y = p[(size_t)pt*3+1], z = p[(size_t)pt*3+2];
  for (int pl = 0; pl < 3; ++pl) {
    const float* C = cmat + (b*3 + pl) * 12;
    float inv = 1.0f / C[9];
    float qx = (C[0]*x + C[1]*y + C[2]*z) * inv;
    float qy = (C[3]*x + C[4]*y + C[5]*z) * inv;
    float ux = qx / 1.101f + 0.5f, uy = qy / 1.101f + 0.5f;
    ux = fminf(fmaxf(ux, 0.f), 1.f - 1e-5f);
    uy = fminf(fmaxf(uy, 0.f), 1.f - 1e-5f);
    int i = (int)(ux * RESO_), j = (int)(uy * RESO_);
    i = i < 0 ? 0 : (i > 63 ? 63 : i);
    j = j < 0 ? 0 : (j > 63 ? 63 : j);
    idx[pt * 3 + pl] = i + RESO_ * j;
  }
}

// ---- fc_pos + block 0 -------------------------------------------------------
__global__ void __launch_bounds__(128)
k_block0(const float* __restrict__ p,
         const _Float16* __restrict__ wfcpos, const float* __restrict__ bfcpos,
         const _Float16* __restrict__ wfc0, const float* __restrict__ bfc0,
         const _Float16* __restrict__ wfc1, const float* __restrict__ bfc1,
         const _Float16* __restrict__ wsc, float* __restrict__ net) {
  __shared__ __align__(32) _Float16 X[4][16][256];
  __shared__ __align__(32) _Float16 A2[4][16][128];
  __shared__ __align__(32) _Float16 P[4][16][32];
  const int wv = threadIdx.x >> 5, lane = threadIdx.x & 31;
  const int nl = lane & 15, mbase = (lane >> 4) * 8;
  const int pt0 = blockIdx.x * 64 + wv * 16;

  { // stage p (pad K=3 -> 32)
    int m = lane & 15, cb = (lane >> 4) * 16;
    const float* pp = p + (size_t)(pt0 + m) * 3;
    for (int c = 0; c < 16; ++c) {
      int k = cb + c;
      P[wv][m][k] = (k < 3) ? (_Float16)pp[k] : (_Float16)0.0f;
    }
  }
  __syncthreads();
  v16h ap = load_afrag(&P[wv][0][0], 32, lane, 0);
  // x = p @ fc_pos + b  (N=256, raw, no relu)
  for (int nt = 0; nt < 16; ++nt) {
    v8f acc = {};
    acc = wmma32(ap, load_bfrag(wfcpos, 1, nt, 0, lane), acc);
    float bias = bfcpos[nt * 16 + nl];
    #pragma unroll
    for (int r = 0; r < 8; ++r) X[wv][mbase + r][nt * 16 + nl] = (_Float16)(acc[r] + bias);
  }
  __syncthreads();
  v16h xf[8];
  #pragma unroll
  for (int k = 0; k < 8; ++k) xf[k] = load_afrag(&X[wv][0][0], 256, lane, 32 * k);
  // n0 = relu( relu(x) @ fc0 + b0 )
  for (int nt = 0; nt < 8; ++nt) {
    v8f acc = {};
    #pragma unroll
    for (int k = 0; k < 8; ++k) acc = wmma32(frag_relu(xf[k]), load_bfrag(wfc0, 8, nt, k, lane), acc);
    float bias = bfc0[nt * 16 + nl];
    #pragma unroll
    for (int r = 0; r < 8; ++r) {
      float v = acc[r] + bias; v = v > 0.f ? v : 0.f;
      A2[wv][mbase + r][nt * 16 + nl] = (_Float16)v;
    }
  }
  __syncthreads();
  v16h nf[4];
  #pragma unroll
  for (int k = 0; k < 4; ++k) nf[k] = load_afrag(&A2[wv][0][0], 128, lane, 32 * k);
  // net = x @ sc + (n0 @ fc1 + b1)
  for (int nt = 0; nt < 8; ++nt) {
    v8f ad = {}, as = {};
    #pragma unroll
    for (int k = 0; k < 4; ++k) ad = wmma32(nf[k], load_bfrag(wfc1, 4, nt, k, lane), ad);
    #pragma unroll
    for (int k = 0; k < 8; ++k) as = wmma32(xf[k], load_bfrag(wsc, 8, nt, k, lane), as);
    float bias = bfc1[nt * 16 + nl];
    #pragma unroll
    for (int r = 0; r < 8; ++r)
      net[(size_t)(pt0 + mbase + r) * 128 + nt * 16 + nl] = as[r] + ad[r] + bias;
  }
}

// ---- bins init + scatter max ------------------------------------------------
__global__ void __launch_bounds__(256)
k_scatter(const float* __restrict__ net, const int* __restrict__ idx, unsigned* __restrict__ bins) {
  int t = blockIdx.x * blockDim.x + threadIdx.x;
  if (t >= B_ * T_ * 8) return;
  int pt = t >> 3, g = t & 7;
  int b = pt / T_;
  const float* nr = net + (size_t)pt * 128 + g * 16;
  size_t base0 = (((size_t)(b*3+0) * R2_) + idx[pt*3+0]) * 128 + g * 16;
  size_t base1 = (((size_t)(b*3+1) * R2_) + idx[pt*3+1]) * 128 + g * 16;
  size_t base2 = (((size_t)(b*3+2) * R2_) + idx[pt*3+2]) * 128 + g * 16;
  for (int c = 0; c < 16; ++c) {
    unsigned key = fkey(nr[c]);
    atomicMax(&bins[base0 + c], key);
    atomicMax(&bins[base1 + c], key);
    atomicMax(&bins[base2 + c], key);
  }
}

// ---- gather pooled + block i (i=1..4) ---------------------------------------
__global__ void __launch_bounds__(128)
k_gblock(float* __restrict__ net, const unsigned* __restrict__ bins, const int* __restrict__ idx,
         const _Float16* __restrict__ wfc0, const float* __restrict__ bfc0,
         const _Float16* __restrict__ wfc1, const float* __restrict__ bfc1,
         const _Float16* __restrict__ wsc) {
  __shared__ __align__(32) _Float16 X[4][16][256];
  __shared__ __align__(32) _Float16 A2[4][16][128];
  __shared__ int Ldx[4][16][3];
  const int wv = threadIdx.x >> 5, lane = threadIdx.x & 31;
  const int nl = lane & 15, mbase = (lane >> 4) * 8;
  const int pt0 = blockIdx.x * 64 + wv * 16;
  const int b = pt0 / T_;

  if (lane < 16) {
    const int* ip = idx + (size_t)(pt0 + lane) * 3;
    Ldx[wv][lane][0] = ip[0]; Ldx[wv][lane][1] = ip[1]; Ldx[wv][lane][2] = ip[2];
  }
  __syncthreads();
  // x = concat(net, pooled) as f16
  for (int s = lane; s < 16 * 128; s += 32) {
    int m = s >> 7, c = s & 127;
    X[wv][m][c] = (_Float16)net[(size_t)(pt0 + m) * 128 + c];
    float pool = 0.f;
    #pragma unroll
    for (int pl = 0; pl < 3; ++pl) {
      unsigned key = bins[(((size_t)(b*3+pl) * R2_) + Ldx[wv][m][pl]) * 128 + c];
      pool += funkey(key);
    }
    X[wv][m][128 + c] = (_Float16)pool;
  }
  __syncthreads();
  v16h xf[8];
  #pragma unroll
  for (int k = 0; k < 8; ++k) xf[k] = load_afrag(&X[wv][0][0], 256, lane, 32 * k);
  for (int nt = 0; nt < 8; ++nt) {
    v8f acc = {};
    #pragma unroll
    for (int k = 0; k < 8; ++k) acc = wmma32(frag_relu(xf[k]), load_bfrag(wfc0, 8, nt, k, lane), acc);
    float bias = bfc0[nt * 16 + nl];
    #pragma unroll
    for (int r = 0; r < 8; ++r) {
      float v = acc[r] + bias; v = v > 0.f ? v : 0.f;
      A2[wv][mbase + r][nt * 16 + nl] = (_Float16)v;
    }
  }
  __syncthreads();
  v16h nf[4];
  #pragma unroll
  for (int k = 0; k < 4; ++k) nf[k] = load_afrag(&A2[wv][0][0], 128, lane, 32 * k);
  for (int nt = 0; nt < 8; ++nt) {
    v8f ad = {}, as = {};
    #pragma unroll
    for (int k = 0; k < 4; ++k) ad = wmma32(nf[k], load_bfrag(wfc1, 4, nt, k, lane), ad);
    #pragma unroll
    for (int k = 0; k < 8; ++k) as = wmma32(xf[k], load_bfrag(wsc, 8, nt, k, lane), as);
    float bias = bfc1[nt * 16 + nl];
    #pragma unroll
    for (int r = 0; r < 8; ++r)
      net[(size_t)(pt0 + mbase + r) * 128 + nt * 16 + nl] = as[r] + ad[r] + bias;
  }
}

// ---- c = net @ fc_c + b + net_pl; scatter-add sums/counts -------------------
__global__ void __launch_bounds__(128)
k_final_c(const float* __restrict__ net, const _Float16* __restrict__ wfcc, const float* __restrict__ bfcc,
          const float* __restrict__ netpl, const int* __restrict__ idx,
          float* __restrict__ sums, float* __restrict__ counts) {
  __shared__ __align__(32) _Float16 X[4][16][128];
  __shared__ int Ldx[4][16][3];
  const int wv = threadIdx.x >> 5, lane = threadIdx.x & 31;
  const int nl = lane & 15, mbase = (lane >> 4) * 8;
  const int pt0 = blockIdx.x * 64 + wv * 16;
  const int b = pt0 / T_;

  if (lane < 16) {
    const int* ip = idx + (size_t)(pt0 + lane) * 3;
    Ldx[wv][lane][0] = ip[0]; Ldx[wv][lane][1] = ip[1]; Ldx[wv][lane][2] = ip[2];
  }
  for (int s = lane; s < 16 * 128; s += 32) {
    int m = s >> 7, c = s & 127;
    X[wv][m][c] = (_Float16)net[(size_t)(pt0 + m) * 128 + c];
  }
  __syncthreads();
  v16h xf[4];
  #pragma unroll
  for (int k = 0; k < 4; ++k) xf[k] = load_afrag(&X[wv][0][0], 128, lane, 32 * k);
  for (int nt = 0; nt < 8; ++nt) {
    v8f acc = {};
    #pragma unroll
    for (int k = 0; k < 4; ++k) acc = wmma32(xf[k], load_bfrag(wfcc, 4, nt, k, lane), acc);
    int n = nt * 16 + nl;
    float add = bfcc[n] + netpl[b * 128 + n];
    #pragma unroll
    for (int r = 0; r < 8; ++r) {
      float cv = acc[r] + add;
      int m = mbase + r;
      #pragma unroll
      for (int pl = 0; pl < 3; ++pl) {
        size_t cell = Ldx[wv][m][pl];
        atomicAdd(&sums[(((size_t)(b*3+pl) * R2_) + cell) * 128 + n], cv);
      }
    }
    if (nt == 0 && nl == 0) {
      #pragma unroll
      for (int r = 0; r < 8; ++r) {
        int m = mbase + r;
        #pragma unroll
        for (int pl = 0; pl < 3; ++pl)
          atomicAdd(&counts[(size_t)(b*3+pl) * R2_ + Ldx[wv][m][pl]], 1.0f);
      }
    }
  }
}

// ---- feas = sums / max(counts,1), transposed to (B,3,128,64,64) -------------
__global__ void __launch_bounds__(256)
k_feat(const float* __restrict__ sums, const float* __restrict__ counts, float* __restrict__ out) {
  size_t o = (size_t)blockIdx.x * blockDim.x + threadIdx.x;
  if (o >= (size_t)B_ * 3 * 128 * R2_) return;
  int cell = (int)(o & (R2_ - 1));
  int ch = (int)((o >> 12) & 127);
  int bp = (int)(o >> 19);
  float s = sums[(((size_t)bp * R2_) + cell) * 128 + ch];
  float c = counts[(size_t)bp * R2_ + cell];
  out[o] = s / fmaxf(c, 1.0f);
}

// ---- host-side orchestration ------------------------------------------------
extern "C" void kernel_launch(void* const* d_in, const int* in_sizes, int n_in,
                              void* d_out, int out_size, void* d_ws, size_t ws_size,
                              hipStream_t stream) {
  const float* p         = (const float*)d_in[0];
  const float* pln_in_w  = (const float*)d_in[1];
  const float* pln_in_b  = (const float*)d_in[2];
  const float* pln_h1_w  = (const float*)d_in[3];
  const float* pln_h1_b  = (const float*)d_in[4];
  const float* pln_h2_w  = (const float*)d_in[5];
  const float* pln_h2_b  = (const float*)d_in[6];
  const float* pln_h3_w  = (const float*)d_in[7];
  const float* pln_h3_b  = (const float*)d_in[8];
  const float* pln_out_w = (const float*)d_in[9];
  const float* pln_out_b = (const float*)d_in[10];
  const float* ph_w      = (const float*)d_in[11];
  const float* ph_b      = (const float*)d_in[12];
  const float* fc_pos_w  = (const float*)d_in[13];
  const float* fc_pos_b  = (const float*)d_in[14];
  const float* blk_fc0_w = (const float*)d_in[15];
  const float* blk_fc0_b = (const float*)d_in[16];
  const float* blk_fc1_w = (const float*)d_in[17];
  const float* blk_fc1_b = (const float*)d_in[18];
  const float* blk_sc_w  = (const float*)d_in[19];
  const float* fc_c_w    = (const float*)d_in[20];
  const float* fc_c_b    = (const float*)d_in[21];

  char* ws = (char*)d_ws;
  _Float16* w16   = (_Float16*)ws;
  float*    hmax  = (float*)(ws + OFF_HMAX);
  float*    netpl = (float*)(ws + OFF_NETPL);
  float*    cmat  = (float*)(ws + OFF_CMAT);
  int*      idx   = (int*)(ws + OFF_IDX);
  float*    net   = (float*)(ws + OFF_NET);
  unsigned* bins  = (unsigned*)(ws + OFF_BINS);
  float*    sums  = (float*)(ws + OFF_BINS);
  float*    counts= (float*)(ws + OFF_CNT);
  float*    out   = (float*)d_out;
  float*    outc  = out + (size_t)B_ * 3 * 128 * R2_;

  auto g = [](long n) { return dim3((unsigned)((n + 255) / 256)); };

  // weight pre-swizzle to WMMA B-fragment layout (f16)
  k_swizzle<<<g(32*128),  256, 0, stream>>>(pln_in_w, w16 + W_PLN_IN, 3, 32, 128);
  k_swizzle<<<g(128*128), 256, 0, stream>>>(pln_h1_w, w16 + W_PLN_H1, 128, 128, 128);
  k_swizzle<<<g(128*128), 256, 0, stream>>>(pln_h2_w, w16 + W_PLN_H2, 128, 128, 128);
  k_swizzle<<<g(32*256),  256, 0, stream>>>(fc_pos_w, w16 + W_FC_POS, 3, 32, 256);
  for (int i = 0; i < 5; ++i) {
    k_swizzle<<<g(256*128), 256, 0, stream>>>(blk_fc0_w + (size_t)i*256*128, w16 + W_FC0 + i*32768, 256, 256, 128);
    k_swizzle<<<g(128*128), 256, 0, stream>>>(blk_fc1_w + (size_t)i*128*128, w16 + W_FC1 + i*16384, 128, 128, 128);
    k_swizzle<<<g(256*128), 256, 0, stream>>>(blk_sc_w  + (size_t)i*256*128, w16 + W_SC  + i*32768, 256, 256, 128);
  }
  k_swizzle<<<g(128*128), 256, 0, stream>>>(fc_c_w, w16 + W_FCC, 128, 128, 128);

  // pointnet branch
  k_fill_f32<<<g(B_*128), 256, 0, stream>>>(hmax, 0.f, (long)B_ * 128);
  k_pointnet<<<dim3(B_*T_/128), 256, 0, stream>>>(p, pln_in_b, pln_h1_b, pln_h2_b,
      w16 + W_PLN_IN, w16 + W_PLN_H1, w16 + W_PLN_H2, hmax);
  k_head<<<dim3(B_), 128, 0, stream>>>(hmax, pln_h3_w, pln_h3_b, pln_out_w, pln_out_b,
      ph_w, ph_b, netpl, cmat, outc);
  k_idx<<<g(B_*T_), 256, 0, stream>>>(p, cmat, idx);

  // trunk
  k_block0<<<dim3(B_*T_/64), 128, 0, stream>>>(p, w16 + W_FC_POS, fc_pos_b,
      w16 + W_FC0, blk_fc0_b, w16 + W_FC1, blk_fc1_b, w16 + W_SC, net);
  for (int i = 1; i < 5; ++i) {
    k_fill_key<<<g((long)B_*3*R2_*128), 256, 0, stream>>>(bins, -1e30f, (long)B_*3*R2_*128);
    k_scatter<<<g((long)B_*T_*8), 256, 0, stream>>>(net, idx, bins);
    k_gblock<<<dim3(B_*T_/64), 128, 0, stream>>>(net, bins, idx,
        w16 + W_FC0 + i*32768, blk_fc0_b + i*128,
        w16 + W_FC1 + i*16384, blk_fc1_b + i*128,
        w16 + W_SC  + i*32768);
  }

  // final projection + mean pooling into planes
  k_fill_f32<<<g((long)B_*3*R2_*128), 256, 0, stream>>>(sums, 0.f, (long)B_*3*R2_*128);
  k_fill_f32<<<g((long)B_*3*R2_), 256, 0, stream>>>(counts, 0.f, (long)B_*3*R2_);
  k_final_c<<<dim3(B_*T_/64), 128, 0, stream>>>(net, w16 + W_FCC, fc_c_b, netpl, idx, sums, counts);
  k_feat<<<g((long)B_*3*128*R2_), 256, 0, stream>>>(sums, counts, out);
}
